// PatchEncoder_57664230916957
// MI455X (gfx1250) — matrix operand order repaired and checked by reading, not compiled, not run on hardware
//
#include <hip/hip_runtime.h>

typedef __attribute__((ext_vector_type(16))) _Float16 v16h;
typedef __attribute__((ext_vector_type(8)))  _Float16 v8h;
typedef __attribute__((ext_vector_type(4)))  _Float16 v4h;
typedef __attribute__((ext_vector_type(8)))  float    v8f;
typedef __attribute__((ext_vector_type(4)))  float    v4f;

#define IMG_H 512
#define IMG_W 512
#define IMG_C 128

#define PSTR 136                    // padded LDS stride (halves) per pixel / weight row
#define XS_HALFS (18*18*PSTR)       // 44064 halves = 88128 B input halo tile
#define WS_HALFS (128*PSTR)         // 17408 halves = 34816 B one tap of weights
// total dynamic LDS: 88128 + 2*34816 = 157760 B

// ---------------------------------------------------------------------------
// Kernel 0: conv_kernel (3,3,Cin,Cout) fp32 HWIO  ->  Wt[tap][co][ci] f16
// ---------------------------------------------------------------------------
__global__ __launch_bounds__(256)
void wprep_kernel(const float* __restrict__ k4, _Float16* __restrict__ wt) {
    int idx = blockIdx.x * 256 + threadIdx.x;   // [0, 9*128*128)
    int tap = idx >> 14;                        // /16384
    int rem = idx & 16383;
    int co  = rem >> 7;
    int ci  = rem & 127;
    // src flat: ((tap)*128 + ci)*128 + co
    wt[idx] = (_Float16)k4[(tap << 14) + (ci << 7) + co];
}

// ---------------------------------------------------------------------------
// Async global->LDS staging of one tap's 128x128 f16 weight slab.
// Each of 512 threads copies 4 x 16B chunks; LDS rows padded to PSTR halves.
// Tracked by ASYNCcnt; caller must s_wait_asynccnt + barrier before use.
// ---------------------------------------------------------------------------
__device__ __forceinline__
void stage_weights_async(const _Float16* __restrict__ wsrc,
                         _Float16* wdst, int tid)
{
    #pragma unroll
    for (int i = 0; i < 4; ++i) {
        int idx = tid + (i << 9);               // 0..2047 16B chunks
        int n   = idx >> 4;                     // co row
        int kv  = idx & 15;                     // 8-half chunk within row
        const _Float16* g = wsrc + (n << 7) + (kv << 3);
        unsigned lds = (unsigned)(uintptr_t)(wdst + n * PSTR + (kv << 3));
        asm volatile("global_load_async_to_lds_b128 %0, %1, off"
                     :: "v"(lds), "v"(g) : "memory");
    }
}

__device__ __forceinline__ void wait_async_all() {
    asm volatile("s_wait_asynccnt 0x0" ::: "memory");
}

// ---------------------------------------------------------------------------
// Kernel 1: fused conv3x3 (WMMA f16->f32) + bias + pos_emb + patch permutation
// One workgroup (512 thr = 16 waves) per 16x16 coarse patch; grid 32x32.
// Weights double-buffered in LDS via async-to-LDS loads.
// ---------------------------------------------------------------------------
__global__ __launch_bounds__(512)
void patchenc_wmma_kernel(const float* __restrict__ X,
                          const float* __restrict__ bias,
                          const float* __restrict__ pos,
                          const _Float16* __restrict__ Wt,
                          float* __restrict__ out)
{
    extern __shared__ __align__(16) _Float16 smem[];
    _Float16* Xs  = smem;                         // [18*18][PSTR] f16 input halo
    _Float16* Ws0 = smem + XS_HALFS;              // weight buffer 0
    _Float16* Ws1 = smem + XS_HALFS + WS_HALFS;   // weight buffer 1

    const int bx   = blockIdx.x;      // coarse patch col (0..31)
    const int by   = blockIdx.y;      // coarse patch row (0..31)
    const int tid  = threadIdx.x;
    const int lane = tid & 31;
    const int wave = tid >> 5;        // 0..15
    const int mg   = wave & 7;        // M-group: rows 2*mg, 2*mg+1 of the patch
    const int ng   = wave >> 3;       // N-group: channels [64*ng, 64*ng+64)
    const int ksel = lane >> 4;       // 0/1 half-wave selector
    const int mlan = lane & 15;

    // ---- kick off tap 0 weight staging (async, into buffer 0) ----
    stage_weights_async(Wt, Ws0, tid);

    // ---- stage 18x18x128 fp32 halo -> f16 LDS (zero-pad at image borders) ----
    for (int idx = tid; idx < 18*18*32; idx += 512) {
        int pix = idx >> 5;           // 0..323
        int cv  = idx & 31;           // float4 index within channels
        int ty  = pix / 18;
        int tx  = pix - ty * 18;
        int h   = (by << 4) - 1 + ty;
        int w   = (bx << 4) - 1 + tx;
        v4f val = {0.f, 0.f, 0.f, 0.f};
        if ((unsigned)h < (unsigned)IMG_H && (unsigned)w < (unsigned)IMG_W)
            val = *(const v4f*)(X + ((((h << 9) + w) << 7) + (cv << 2)));
        v4h hv = { (_Float16)val.x, (_Float16)val.y, (_Float16)val.z, (_Float16)val.w };
        *(v4h*)(Xs + pix * PSTR + (cv << 2)) = hv;
    }

    wait_async_all();
    __syncthreads();                  // Xs + Ws0 ready for all waves

    v8f acc[2][4] = {};   // 2 M-tiles x 4 N-tiles of 16x16 f32 accumulators

    for (int tap = 0; tap < 9; ++tap) {
        _Float16* Wsc = (tap & 1) ? Ws1 : Ws0;

        // issue next tap's staging into the other buffer (WAR-safe: its last
        // readers finished before the barrier that ended iteration tap-1)
        if (tap < 8)
            stage_weights_async(Wt + ((tap + 1) << 14),
                                (tap & 1) ? Ws0 : Ws1, tid);

        const int dy = tap / 3;
        const int dx = tap - dy * 3;

        #pragma unroll
        for (int kc = 0; kc < 4; ++kc) {          // 4 x K=32 chunks of Cin=128
            v16h a[2], b[4];
            #pragma unroll
            for (int mi = 0; mi < 2; ++mi) {
                int mt  = (mg << 1) + mi;          // patch row = A tile
                int pix = (mt + dy) * 18 + mlan + dx;
                const _Float16* p = Xs + pix * PSTR + (kc << 5) + (ksel << 3);
                v8h lo = *(const v8h*)p;
                v8h hi = *(const v8h*)(p + 16);
                a[mi] = __builtin_shufflevector(lo, hi,
                        0,1,2,3,4,5,6,7,8,9,10,11,12,13,14,15);
            }
            #pragma unroll
            for (int ni = 0; ni < 4; ++ni) {
                int nt = (ng << 2) + ni;
                const _Float16* p = Wsc + ((nt << 4) + mlan) * PSTR
                                        + (kc << 5) + (ksel << 3);
            v8h lo = *(const v8h*)p;
            v8h hi = *(const v8h*)(p + 16);
                b[ni] = __builtin_shufflevector(lo, hi,
                        0,1,2,3,4,5,6,7,8,9,10,11,12,13,14,15);
            }
            #pragma unroll
            for (int mi = 0; mi < 2; ++mi)
                #pragma unroll
                for (int ni = 0; ni < 4; ++ni)
                    acc[mi][ni] = __builtin_amdgcn_wmma_f32_16x16x32_f16(
                        false, a[mi], false, b[ni],
                        (short)0, acc[mi][ni], false, false);
        }

        // own async ops complete, then cross-wave visibility for next tap
        wait_async_all();
        __syncthreads();
    }

    // ---- epilogue: + bias + pos_emb, store to contiguous coarse patch ----
    float* outp = out + (((long long)((by << 5) + bx)) << 15);   // *32768
    #pragma unroll
    for (int mi = 0; mi < 2; ++mi) {
        const int py = (mg << 1) + mi;                 // row in 16x16 patch
        #pragma unroll
        for (int ni = 0; ni < 4; ++ni) {
            const int nt = (ng << 2) + ni;
            const int c  = (nt << 4) + mlan;           // output channel
            const float bv = bias[c];
            v8f d = acc[mi][ni];
            #pragma unroll
            for (int v = 0; v < 8; ++v) {
                const int px = v + (ksel << 3);        // col in 16x16 patch
                const int nf = ((by << 2) + (py >> 2)) * 128 + (bx << 2) + (px >> 2);
                const int df = (((py & 3) << 2) + (px & 3)) * 128 + c;
                const float pv = pos[(long long)nf * 2048 + df];
                outp[(((py << 4) + px) << 7) + c] = d[v] + bv + pv;
            }
        }
    }
}

// ---------------------------------------------------------------------------
extern "C" void kernel_launch(void* const* d_in, const int* in_sizes, int n_in,
                              void* d_out, int out_size, void* d_ws, size_t ws_size,
                              hipStream_t stream) {
    (void)in_sizes; (void)n_in; (void)out_size; (void)ws_size;
    const float* X    = (const float*)d_in[0];   // [1,512,512,128]
    const float* ck   = (const float*)d_in[1];   // [3,3,128,128]
    const float* bias = (const float*)d_in[2];   // [128]
    const float* pos  = (const float*)d_in[3];   // [16384,2048]
    float* out = (float*)d_out;                  // [1,1024,32768]
    _Float16* Wt = (_Float16*)d_ws;              // 9*128*128 f16 = 288 KB

    wprep_kernel<<<576, 256, 0, stream>>>(ck, Wt);

    dim3 grid(32, 32);
    size_t lds_bytes = (size_t)(XS_HALFS + 2 * WS_HALFS) * sizeof(_Float16); // 157760
    patchenc_wmma_kernel<<<grid, 512, lds_bytes, stream>>>(X, bias, pos, Wt, out);
}
